// SlidingWindowMultiheadAttention_41205916238373
// MI455X (gfx1250) — matrix operand order repaired and checked by reading, not compiled, not run on hardware
//
#include <hip/hip_runtime.h>
#include <math.h>

typedef __attribute__((ext_vector_type(16))) __bf16 v16bf;
typedef __attribute__((ext_vector_type(8)))  __bf16 v8bf;
typedef __attribute__((ext_vector_type(8)))  float  v8f;

static constexpr int Bsz = 4;
static constexpr int S   = 2048;
static constexpr int D   = 512;
static constexpr int H   = 8;
static constexpr int Dh  = 64;
static constexpr int M   = Bsz * S;  // 8192 rows

__device__ __forceinline__ v8f zero8() {
  v8f z;
#pragma unroll
  for (int i = 0; i < 8; ++i) z[i] = 0.0f;
  return z;
}

// A-operand (16x32, 16-bit): lane holds row l%16; K chunks {kb..kb+7} U {kb+16..kb+23}, kb = 8*(lane>=16)
__device__ __forceinline__ v16bf load_a(const __bf16* row, int kbase) {
  union { v16bf v; v8bf h[2]; } u;
  u.h[0] = *reinterpret_cast<const v8bf*>(row + kbase);
  u.h[1] = *reinterpret_cast<const v8bf*>(row + kbase + 16);
  return u.v;
}

// B-operand (32x16, 16-bit): lane holds column l%16 == a source row; K chunks {8*half..8*half+7} U {16+8*half..16+8*half+7}
__device__ __forceinline__ v16bf load_b(const __bf16* row, int half) {
  union { v16bf v; v8bf h[2]; } u;
  u.h[0] = *reinterpret_cast<const v8bf*>(row + half * 8);
  u.h[1] = *reinterpret_cast<const v8bf*>(row + half * 8 + 16);
  return u.v;
}

__device__ __forceinline__ v8f wmma_bf16(v16bf a, v16bf b, v8f c) {
  return __builtin_amdgcn_wmma_f32_16x16x32_bf16(false, a, false, b, (short)0, c, false, false);
}

// ---------------- fp32 -> bf16 convert ----------------
__global__ void cvt_bf16_kernel(const float* __restrict__ in, __bf16* __restrict__ out, int n) {
  int i = blockIdx.x * blockDim.x + threadIdx.x;
  if (i < n) out[i] = (__bf16)in[i];
}

// ---------------- 32x64-per-wave GEMM core: C = A(bf16) * W(bf16)^T ----------------
// Unrolled x4: the scheduler renames per-slice fragment registers and hoists the
// load clauses across earlier slices' WMMAs (no explicit ping-pong, no v_mov rotation).
__device__ __forceinline__ void gemm_32x64(const __bf16* __restrict__ A,
                                           const __bf16* __restrict__ W,
                                           int m0, int n0, v8f acc[2][4],
                                           int l16, int half, int kbase) {
  const __bf16* arow0 = A + (size_t)(m0 + l16) * D;
  const __bf16* arow1 = arow0 + (size_t)16 * D;
  const __bf16* brow0 = W + (size_t)(n0 + l16) * D;

#pragma unroll 4
  for (int k = 0; k < D; k += 32) {
    v16bf a0 = load_a(arow0 + k, kbase);
    v16bf a1 = load_a(arow1 + k, kbase);
    v16bf b[4];
#pragma unroll
    for (int t = 0; t < 4; ++t) b[t] = load_b(brow0 + (size_t)t * 16 * D + k, half);
#pragma unroll
    for (int t = 0; t < 4; ++t) {
      acc[0][t] = wmma_bf16(a0, b[t], acc[0][t]);
      acc[1][t] = wmma_bf16(a1, b[t], acc[1][t]);
    }
  }
}

// ---------------- fused QKV projection ----------------
// z=0: Q [B,H,S,Dh] (scaled by 1/sqrt(Dh));  z=1: K [B,H,S,Dh];  z=2: V^T [B,H,Dh,S]
__global__ __launch_bounds__(256) void qkv_gemm_kernel(
    const __bf16* __restrict__ X,
    const __bf16* __restrict__ Wq, const __bf16* __restrict__ Wk, const __bf16* __restrict__ Wv,
    __bf16* __restrict__ Q, __bf16* __restrict__ Kb, __bf16* __restrict__ Vt) {
  const int lane = threadIdx.x & 31, wv = threadIdx.x >> 5;
  const int half = lane >> 4, l16 = lane & 15, kbase = half * 8;
  const int m0 = (blockIdx.x * 8 + wv) * 32;
  const int n0 = blockIdx.y * 64;
  const int z  = blockIdx.z;
  const __bf16* W = (z == 0) ? Wq : ((z == 1) ? Wk : Wv);

  v8f acc[2][4];
#pragma unroll
  for (int i = 0; i < 2; ++i)
#pragma unroll
    for (int t = 0; t < 4; ++t) acc[i][t] = zero8();

  gemm_32x64(X, W, m0, n0, acc, l16, half, kbase);

  const float sc = (z == 0) ? 0.125f : 1.0f;  // 1/sqrt(64)
#pragma unroll
  for (int i = 0; i < 2; ++i) {
#pragma unroll
    for (int t = 0; t < 4; ++t) {
      const int n = n0 + t * 16 + l16, h = n >> 6, d = n & 63;
#pragma unroll
      for (int j = 0; j < 8; ++j) {
        const int m = m0 + i * 16 + j + half * 8;
        const int b = m >> 11, s = m & (S - 1);
        const float v = acc[i][t][j] * sc;
        if (z == 2)
          Vt[(((size_t)(b * H + h)) * Dh + d) * S + s] = (__bf16)v;
        else if (z == 0)
          Q[(((size_t)(b * H + h)) * S + s) * Dh + d] = (__bf16)v;
        else
          Kb[(((size_t)(b * H + h)) * S + s) * Dh + d] = (__bf16)v;
      }
    }
  }
}

// ---------------- flash attention over the banded key range ----------------
__global__ __launch_bounds__(256) void attn_kernel(
    const __bf16* __restrict__ Q, const __bf16* __restrict__ Kb, const __bf16* __restrict__ Vt,
    const int* __restrict__ amask, const int* __restrict__ winp, __bf16* __restrict__ Obuf) {
  __shared__ __align__(16) __bf16 Plds[8][16][32];  // per-wave P tile, row-major [query][key]

  const int lane = threadIdx.x & 31, wv = threadIdx.x >> 5;
  const int half = lane >> 4, l16 = lane & 15, kbase = half * 8;
  const int h = blockIdx.y, b = blockIdx.z;
  const int bh = b * H + h;
  const int q0 = blockIdx.x * 128 + wv * 16;
  const int window = *winp;

  // Q fragments (A-operand), Dh split into two 32-wide K slices
  const __bf16* qrow = Q + ((size_t)bh * S + q0 + l16) * Dh;
  const v16bf a_lo = load_a(qrow, kbase);
  const v16bf a_hi = load_a(qrow + 32, kbase);

  v8f acc[4] = {zero8(), zero8(), zero8(), zero8()};
  float ms[8], ls[8];
#pragma unroll
  for (int j = 0; j < 8; ++j) { ms[j] = -__builtin_inff(); ls[j] = 0.0f; }

  int kend = q0 + 15 + window;          // last allowed key + 1 for this wave's max query
  if (kend > S) kend = S;
  kend = (kend + 31) & ~31;
  if (kend > S) kend = S;

  const __bf16* kmat = Kb + (size_t)bh * S * Dh;
  const __bf16* vmat = Vt + (size_t)bh * Dh * S;

  auto loadK = [&](int kt, v16bf kb[4]) {
    const __bf16* r0 = kmat + (size_t)(kt + l16) * Dh;
    const __bf16* r1 = r0 + (size_t)16 * Dh;
    kb[0] = load_b(r0, half);        // keys kt..kt+15, dh 0..31
    kb[1] = load_b(r0 + 32, half);   // keys kt..kt+15, dh 32..63
    kb[2] = load_b(r1, half);        // keys kt+16..kt+31, dh 0..31
    kb[3] = load_b(r1 + 32, half);   // keys kt+16..kt+31, dh 32..63
  };

  // one 32-key tile: scores -> online softmax -> LDS transpose of P -> O += P*V
  auto tile_body = [&](int kt, const v16bf kb[4]) {
    v8f s0 = zero8(), s1 = zero8();
    s0 = wmma_bf16(a_lo, kb[0], s0);
    s0 = wmma_bf16(a_hi, kb[1], s0);
    s1 = wmma_bf16(a_lo, kb[2], s1);
    s1 = wmma_bf16(a_hi, kb[3], s1);

    // issue this tile's V fragments now — they overlap the softmax VALU below
    v16bf vf[4];
#pragma unroll
    for (int t = 0; t < 4; ++t) {
      const __bf16* vrow = vmat + (size_t)(t * 16 + l16) * S + kt;
      vf[t] = load_b(vrow, half);
    }

    const int key0 = kt + l16, key1 = kt + 16 + l16;
    const bool mok0 = amask[(size_t)b * S + key0] != 0;
    const bool mok1 = amask[(size_t)b * S + key1] != 0;

    // online softmax per row (row j+8*half lives across 16 lanes of slot j)
#pragma unroll
    for (int j = 0; j < 8; ++j) {
      const int q = q0 + j + half * 8;
      float v0 = ((key0 - q) <= (window - 1) && mok0) ? s0[j] : -__builtin_inff();
      float v1 = ((key1 - q) <= (window - 1) && mok1) ? s1[j] : -__builtin_inff();
      float rm = fmaxf(v0, v1);
      rm = fmaxf(rm, __shfl_xor(rm, 1, 32));
      rm = fmaxf(rm, __shfl_xor(rm, 2, 32));
      rm = fmaxf(rm, __shfl_xor(rm, 4, 32));
      rm = fmaxf(rm, __shfl_xor(rm, 8, 32));
      const float nm = fmaxf(ms[j], rm);
      const float p0 = __expf(v0 - nm);
      const float p1 = __expf(v1 - nm);
      float rs = p0 + p1;
      rs += __shfl_xor(rs, 1, 32);
      rs += __shfl_xor(rs, 2, 32);
      rs += __shfl_xor(rs, 4, 32);
      rs += __shfl_xor(rs, 8, 32);
      const float f = __expf(ms[j] - nm);
      ls[j] = ls[j] * f + rs;
      ms[j] = nm;
#pragma unroll
      for (int t = 0; t < 4; ++t) acc[t][j] *= f;
      const int r = j + half * 8;
      Plds[wv][r][l16]      = (__bf16)p0;
      Plds[wv][r][16 + l16] = (__bf16)p1;
    }
    // cross-lane LDS transpose hazard: same-wave DS is in-order, wait for stores
    asm volatile("s_wait_dscnt 0x0" ::: "memory");

    // O += P * V : P as A-operand (16x32), V^T rows as B-operand
    const v16bf pf = load_a(&Plds[wv][l16][0], kbase);
#pragma unroll
    for (int t = 0; t < 4; ++t) acc[t] = wmma_bf16(pf, vf[t], acc[t]);
  };

  // key loop, unrolled x2 with textual ping-pong buffers (no register rotation)
  v16bf kbA[4], kbB[4];
  loadK(0, kbA);
  int kt = 0;
  for (; kt + 64 <= kend; kt += 64) {
    loadK(kt + 32, kbB);            // prefetch next tile into the idle buffer
    tile_body(kt, kbA);
    if (kt + 64 < kend) loadK(kt + 64, kbA);
    tile_body(kt + 32, kbB);
  }
  if (kt < kend) tile_body(kt, kbA);  // odd tail (kbA prefetched by the guard above)

  // ---- normalize and write [B,S,H*Dh] bf16 for the output projection ----
#pragma unroll
  for (int t = 0; t < 4; ++t) {
    const int d = t * 16 + l16;
#pragma unroll
    for (int j = 0; j < 8; ++j) {
      const int s = q0 + j + half * 8;
      const float inv = 1.0f / ls[j];
      Obuf[((size_t)b * S + s) * D + h * Dh + d] = (__bf16)(acc[t][j] * inv);
    }
  }
}

// ---------------- output projection -> fp32 ----------------
__global__ __launch_bounds__(256) void out_gemm_kernel(
    const __bf16* __restrict__ A, const __bf16* __restrict__ Wo, float* __restrict__ Y) {
  const int lane = threadIdx.x & 31, wv = threadIdx.x >> 5;
  const int half = lane >> 4, l16 = lane & 15, kbase = half * 8;
  const int m0 = (blockIdx.x * 8 + wv) * 32;
  const int n0 = blockIdx.y * 64;

  v8f acc[2][4];
#pragma unroll
  for (int i = 0; i < 2; ++i)
#pragma unroll
    for (int t = 0; t < 4; ++t) acc[i][t] = zero8();

  gemm_32x64(A, Wo, m0, n0, acc, l16, half, kbase);

#pragma unroll
  for (int i = 0; i < 2; ++i) {
#pragma unroll
    for (int t = 0; t < 4; ++t) {
      const int n = n0 + t * 16 + l16;
#pragma unroll
      for (int j = 0; j < 8; ++j) {
        const int m = m0 + i * 16 + j + half * 8;
        Y[(size_t)m * D + n] = acc[i][t][j];
      }
    }
  }
}

extern "C" void kernel_launch(void* const* d_in, const int* in_sizes, int n_in,
                              void* d_out, int out_size, void* d_ws, size_t ws_size,
                              hipStream_t stream) {
  const float* hidden = (const float*)d_in[0];
  const int*   amask  = (const int*)d_in[1];
  const float* Wq     = (const float*)d_in[2];
  const float* Wk     = (const float*)d_in[3];
  const float* Wv     = (const float*)d_in[4];
  const float* Wo     = (const float*)d_in[5];
  const int*   winp   = (const int*)d_in[6];
  float* out = (float*)d_out;

  // workspace layout (bf16 buffers)
  char* ws = (char*)d_ws;
  const size_t szX = (size_t)M * D * 2;             // 8 MiB
  const size_t szW = (size_t)D * D * 2;             // 512 KiB
  const size_t szQ = (size_t)Bsz * H * S * Dh * 2;  // 8 MiB
  __bf16* Xb  = (__bf16*)(ws);
  __bf16* Wqb = (__bf16*)(ws + szX);
  __bf16* Wkb = (__bf16*)(ws + szX + szW);
  __bf16* Wvb = (__bf16*)(ws + szX + 2 * szW);
  __bf16* Wob = (__bf16*)(ws + szX + 3 * szW);
  __bf16* Qb  = (__bf16*)(ws + szX + 4 * szW);
  __bf16* Kbb = (__bf16*)(ws + szX + 4 * szW + szQ);
  __bf16* Vtb = (__bf16*)(ws + szX + 4 * szW + 2 * szQ);
  __bf16* Ob  = (__bf16*)(ws + szX + 4 * szW + 3 * szQ);

  const int nX = M * D, nW = D * D;
  cvt_bf16_kernel<<<(nX + 255) / 256, 256, 0, stream>>>(hidden, Xb, nX);
  cvt_bf16_kernel<<<(nW + 255) / 256, 256, 0, stream>>>(Wq, Wqb, nW);
  cvt_bf16_kernel<<<(nW + 255) / 256, 256, 0, stream>>>(Wk, Wkb, nW);
  cvt_bf16_kernel<<<(nW + 255) / 256, 256, 0, stream>>>(Wv, Wvb, nW);
  cvt_bf16_kernel<<<(nW + 255) / 256, 256, 0, stream>>>(Wo, Wob, nW);

  dim3 gQKV(M / 32 / 8, D / 64, 3);   // 32 x 8 x 3
  qkv_gemm_kernel<<<gQKV, 256, 0, stream>>>(Xb, Wqb, Wkb, Wvb, Qb, Kbb, Vtb);

  dim3 gAtt(S / 128, H, Bsz);         // 16 x 8 x 4
  attn_kernel<<<gAtt, 256, 0, stream>>>(Qb, Kbb, Vtb, amask, winp, Ob);

  dim3 gOut(M / 32 / 8, D / 64, 1);   // 32 x 8
  out_gemm_kernel<<<gOut, 256, 0, stream>>>(Ob, Wob, out);
}
